// EFC_GNN_42511586296345
// MI455X (gfx1250) — matrix operand order repaired and checked by reading, not compiled, not run on hardware
//
#include <hip/hip_runtime.h>

#define NNODES 20000
#define NEDGES 160000
#define ETOT   (NEDGES + NNODES)   // self-loops appended
#define HEADS  4
#define HID    128
#define NEG_SLOPE 0.2f
#define LN_EPS 1e-5f

typedef __bf16 bf16;
typedef __attribute__((ext_vector_type(8)))  __bf16 v8bf;
typedef __attribute__((ext_vector_type(16))) __bf16 v16bf;
typedef __attribute__((ext_vector_type(8)))  float  v8f;

// ---------------- helpers ----------------
__global__ void k_cvt_bf16(const float* __restrict__ in, bf16* __restrict__ out, int n) {
  int i = blockIdx.x * blockDim.x + threadIdx.x;
  if (i < n) out[i] = (bf16)in[i];
}

// W[K,N] row-major -> Wt[N,K] bf16 (column-major view for contiguous B fragments)
__global__ void k_tcvt_bf16(const float* __restrict__ W, bf16* __restrict__ Wt, int K, int N) {
  int i = blockIdx.x * blockDim.x + threadIdx.x;
  if (i >= K * N) return;
  int k = i / N, n = i % N;
  Wt[(size_t)n * K + k] = (bf16)W[i];
}

__global__ void k_zero(float* __restrict__ p, int n) {
  int i = blockIdx.x * blockDim.x + threadIdx.x;
  if (i < n) p[i] = 0.f;
}

// ---------------- WMMA GEMM: D[M,N] = A[M,K] x B[K,N], B given transposed Bt[N,K]
// one wave -> 16x64 output strip (4 accumulators), 4 waves per block.
// Double-buffered K loop: next k-step's A/B fragments are in flight while the
// current 4 WMMAs issue, so the backend can use partial s_wait_loadcnt.
// Optionally also emits a bf16 copy of D (fused activation conversion).
__global__ __launch_bounds__(128) void k_wmma_gemm(
    const bf16* __restrict__ A, const bf16* __restrict__ Bt,
    const float* __restrict__ bias, float* __restrict__ D, bf16* __restrict__ Dbf,
    int M, int K, int N, int fuse_bias_relu)
{
  const int lane = threadIdx.x & 31;
  const int wid  = blockIdx.x * 4 + (threadIdx.x >> 5);
  const int nstrips = N >> 6;
  const int mt = wid / nstrips;
  if (mt * 16 >= M) return;
  const int n0 = (wid % nstrips) << 6;
  const int m0 = mt << 4;
  const int hA = (lane >> 4) << 3;   // 0 or 8  (A lane K-offset per ISA layout)
  const int hB = (lane >> 4) << 3;   // 0 or 8  (B lane K-offset per ISA layout)
  const bf16* ap    = A + (size_t)(m0 + (lane & 15)) * K + hA;
  const bf16* bbase = Bt + (size_t)(n0 + (lane & 15)) * K + hB;

  v8f zero = {0.f,0.f,0.f,0.f,0.f,0.f,0.f,0.f};
  v8f acc[4] = {zero, zero, zero, zero};

  // prologue: fragments for kk = 0
  v8bf alo = *(const v8bf*)(ap);
  v8bf ahi = *(const v8bf*)(ap + 16);
  v8bf bl[4], bh[4];
  #pragma unroll
  for (int t = 0; t < 4; ++t) {
    const bf16* bp = bbase + (size_t)t * 16 * K;
    bl[t] = *(const v8bf*)(bp);
    bh[t] = *(const v8bf*)(bp + 16);
  }

  int kk = 0;
  for (; kk + 32 < K; kk += 32) {
    // prefetch next k-step
    v8bf alo_n = *(const v8bf*)(ap + kk + 32);
    v8bf ahi_n = *(const v8bf*)(ap + kk + 48);
    v8bf bl_n[4], bh_n[4];
    #pragma unroll
    for (int t = 0; t < 4; ++t) {
      const bf16* bp = bbase + (size_t)t * 16 * K + kk + 32;
      bl_n[t] = *(const v8bf*)(bp);
      bh_n[t] = *(const v8bf*)(bp + 16);
    }
    // consume current fragments
    v16bf a = __builtin_shufflevector(alo, ahi, 0,1,2,3,4,5,6,7,8,9,10,11,12,13,14,15);
    #pragma unroll
    for (int t = 0; t < 4; ++t) {
      v16bf b = __builtin_shufflevector(bl[t], bh[t], 0,1,2,3,4,5,6,7,8,9,10,11,12,13,14,15);
      acc[t] = __builtin_amdgcn_wmma_f32_16x16x32_bf16(
          false, a, false, b, (short)0, acc[t], false, false);
    }
    alo = alo_n; ahi = ahi_n;
    #pragma unroll
    for (int t = 0; t < 4; ++t) { bl[t] = bl_n[t]; bh[t] = bh_n[t]; }
  }
  // epilogue k-step
  {
    v16bf a = __builtin_shufflevector(alo, ahi, 0,1,2,3,4,5,6,7,8,9,10,11,12,13,14,15);
    #pragma unroll
    for (int t = 0; t < 4; ++t) {
      v16bf b = __builtin_shufflevector(bl[t], bh[t], 0,1,2,3,4,5,6,7,8,9,10,11,12,13,14,15);
      acc[t] = __builtin_amdgcn_wmma_f32_16x16x32_bf16(
          false, a, false, b, (short)0, acc[t], false, false);
    }
  }

  // C/D layout: VGPR r -> row m0 + (lane>=16 ? 8 : 0) + r, col = lane&15
  const int lr = m0 + ((lane >> 4) << 3);
  #pragma unroll
  for (int t = 0; t < 4; ++t) {
    const int col = n0 + t * 16 + (lane & 15);
    float bv = fuse_bias_relu ? bias[col] : 0.f;
    #pragma unroll
    for (int r = 0; r < 8; ++r) {
      float v = acc[t][r] + bv;
      if (fuse_bias_relu) v = fmaxf(v, 0.f);
      D[(size_t)(lr + r) * N + col] = v;
      if (Dbf) Dbf[(size_t)(lr + r) * N + col] = (bf16)v;
    }
  }
}

// ---------------- attention scores per (node, head) ----------------
__global__ void k_att_scores(const float* __restrict__ lin,
                             const float* __restrict__ att_src,
                             const float* __restrict__ att_dst,
                             float* __restrict__ a_s, float* __restrict__ a_d, int C)
{
  int i = blockIdx.x * blockDim.x + threadIdx.x;
  if (i >= NNODES * HEADS) return;
  int n = i >> 2, h = i & 3;
  const float* row = lin + (size_t)n * (HEADS * C) + h * C;
  const float* ps = att_src + h * C;
  const float* pd = att_dst + h * C;
  float s = 0.f, d = 0.f;
  for (int c = 0; c < C; ++c) { float v = row[c]; s = fmaf(v, ps[c], s); d = fmaf(v, pd[c], d); }
  a_s[i] = s; a_d[i] = d;
}

// monotone float<->uint map for atomicMax-based segment max
__device__ __forceinline__ unsigned ordenc(float f) {
  unsigned u = __float_as_uint(f);
  return (u & 0x80000000u) ? ~u : (u | 0x80000000u);
}
__device__ __forceinline__ float orddec(unsigned u) {
  unsigned b = (u & 0x80000000u) ? (u & 0x7fffffffu) : ~u;
  return __uint_as_float(b);
}
__device__ __forceinline__ void edge_sd(const int* __restrict__ ei, int e, int& s, int& d) {
  if (e < NEDGES) { s = ei[e]; d = ei[NEDGES + e]; }
  else            { s = d = e - NEDGES; }          // self-loop
}
__device__ __forceinline__ float lrelu(float a) { return a > 0.f ? a : NEG_SLOPE * a; }

__global__ void k_edge_max(const int* __restrict__ ei, const float* __restrict__ a_s,
                           const float* __restrict__ a_d, unsigned* __restrict__ amax)
{
  int e = blockIdx.x * blockDim.x + threadIdx.x;
  if (e >= ETOT) return;
  int s, d; edge_sd(ei, e, s, d);
  float4 as = *(const float4*)(a_s + 4 * (size_t)s);
  float4 ad = *(const float4*)(a_d + 4 * (size_t)d);
  atomicMax(&amax[4 * d + 0], ordenc(lrelu(as.x + ad.x)));
  atomicMax(&amax[4 * d + 1], ordenc(lrelu(as.y + ad.y)));
  atomicMax(&amax[4 * d + 2], ordenc(lrelu(as.z + ad.z)));
  atomicMax(&amax[4 * d + 3], ordenc(lrelu(as.w + ad.w)));
}

__global__ void k_edge_exp(const int* __restrict__ ei, const float* __restrict__ a_s,
                           const float* __restrict__ a_d, const unsigned* __restrict__ amax,
                           float* __restrict__ denom, float* __restrict__ ex)
{
  int e = blockIdx.x * blockDim.x + threadIdx.x;
  if (e >= ETOT) return;
  int s, d; edge_sd(ei, e, s, d);
  float4 as = *(const float4*)(a_s + 4 * (size_t)s);
  float4 ad = *(const float4*)(a_d + 4 * (size_t)d);
  uint4  am = *(const uint4*)(amax + 4 * (size_t)d);
  float e0 = __expf(lrelu(as.x + ad.x) - orddec(am.x));
  float e1 = __expf(lrelu(as.y + ad.y) - orddec(am.y));
  float e2 = __expf(lrelu(as.z + ad.z) - orddec(am.z));
  float e3 = __expf(lrelu(as.w + ad.w) - orddec(am.w));
  *(float4*)(ex + 4 * (size_t)e) = make_float4(e0, e1, e2, e3);
  atomicAdd(&denom[4 * d + 0], e0);
  atomicAdd(&denom[4 * d + 1], e1);
  atomicAdd(&denom[4 * d + 2], e2);
  atomicAdd(&denom[4 * d + 3], e3);
}

// concat layers (C=32): one thread per (edge, h*C+c), lin stride = 128
__global__ void k_edge_aggr_concat(const int* __restrict__ ei, const float* __restrict__ ex,
                                   const float* __restrict__ denom, const float* __restrict__ lin,
                                   float* __restrict__ agg)
{
  int i = blockIdx.x * blockDim.x + threadIdx.x;
  if (i >= ETOT * 128) return;
  int e = i >> 7, idx = i & 127, h = idx >> 5;
  int s, d; edge_sd(ei, e, s, d);
  float w = ex[4 * e + h] / denom[4 * d + h];
  atomicAdd(&agg[(size_t)d * HID + idx], w * lin[(size_t)s * HID + idx]);
}

// final layer (C=128, mean over heads fused): lin stride = 512, agg stride = 128
__global__ void k_edge_aggr_mean(const int* __restrict__ ei, const float* __restrict__ ex,
                                 const float* __restrict__ denom, const float* __restrict__ lin,
                                 float* __restrict__ agg)
{
  int i = blockIdx.x * blockDim.x + threadIdx.x;
  if (i >= ETOT * 128) return;
  int e = i >> 7, c = i & 127;
  int s, d; edge_sd(ei, e, s, d);
  float4 exv = *(const float4*)(ex + 4 * (size_t)e);
  float4 dn  = *(const float4*)(denom + 4 * (size_t)d);
  const float* ls = lin + (size_t)s * 512;
  float v = exv.x / dn.x * ls[c]       + exv.y / dn.y * ls[128 + c]
          + exv.z / dn.z * ls[256 + c] + exv.w / dn.w * ls[384 + c];
  atomicAdd(&agg[(size_t)d * HID + c], 0.25f * v);
}

// bias + layernorm + optional relu + optional residual; one wave32 per node row.
// Optionally emits bf16 copy of the output row (feeds next layer's WMMA GEMM).
__global__ __launch_bounds__(256) void k_finalize(
    const float* __restrict__ agg, const float* __restrict__ bias,
    const float* __restrict__ lns, const float* __restrict__ lnb,
    const float* __restrict__ h_in, float* __restrict__ h_out, bf16* __restrict__ h_bf,
    int do_relu, int residual)
{
  int n = blockIdx.x * 8 + (threadIdx.x >> 5);
  int lane = threadIdx.x & 31;
  if (n >= NNODES) return;
  float4 v = *(const float4*)(agg + (size_t)n * HID + lane * 4);
  float4 b = *(const float4*)(bias + lane * 4);
  float x0 = v.x + b.x, x1 = v.y + b.y, x2 = v.z + b.z, x3 = v.w + b.w;
  float s = x0 + x1 + x2 + x3;
  float q = x0 * x0 + x1 * x1 + x2 * x2 + x3 * x3;
  #pragma unroll
  for (int m = 16; m >= 1; m >>= 1) { s += __shfl_xor(s, m, 32); q += __shfl_xor(q, m, 32); }
  float mu  = s * (1.f / HID);
  float var = q * (1.f / HID) - mu * mu;
  float r = rsqrtf(var + LN_EPS);
  float4 sc = *(const float4*)(lns + lane * 4);
  float4 bb = *(const float4*)(lnb + lane * 4);
  float o0 = (x0 - mu) * r * sc.x + bb.x;
  float o1 = (x1 - mu) * r * sc.y + bb.y;
  float o2 = (x2 - mu) * r * sc.z + bb.z;
  float o3 = (x3 - mu) * r * sc.w + bb.w;
  if (do_relu) { o0 = fmaxf(o0, 0.f); o1 = fmaxf(o1, 0.f); o2 = fmaxf(o2, 0.f); o3 = fmaxf(o3, 0.f); }
  if (residual) {
    float4 hp = *(const float4*)(h_in + (size_t)n * HID + lane * 4);
    o0 += hp.x; o1 += hp.y; o2 += hp.z; o3 += hp.w;
  }
  *(float4*)(h_out + (size_t)n * HID + lane * 4) = make_float4(o0, o1, o2, o3);
  if (h_bf) {
    bf16* p = h_bf + (size_t)n * HID + lane * 4;
    p[0] = (bf16)o0; p[1] = (bf16)o1; p[2] = (bf16)o2; p[3] = (bf16)o3;
  }
}

// ---------------- host orchestration ----------------
static inline int cdiv(int a, int b) { return (a + b - 1) / b; }

extern "C" void kernel_launch(void* const* d_in, const int* in_sizes, int n_in,
                              void* d_out, int out_size, void* d_ws, size_t ws_size,
                              hipStream_t stream)
{
  const float* x      = (const float*)d_in[0];
  const int*   ei     = (const int*)  d_in[1];
  const float* proj_w = (const float*)d_in[2];
  const float* proj_b = (const float*)d_in[3];
  const float* lin_w[3] = {(const float*)d_in[4],  (const float*)d_in[5],  (const float*)d_in[6]};
  const float* att_s[3] = {(const float*)d_in[7],  (const float*)d_in[8],  (const float*)d_in[9]};
  const float* att_d[3] = {(const float*)d_in[10], (const float*)d_in[11], (const float*)d_in[12]};
  const float* bias[3]  = {(const float*)d_in[13], (const float*)d_in[14], (const float*)d_in[15]};
  const float* ln_s[3]  = {(const float*)d_in[16], (const float*)d_in[17], (const float*)d_in[18]};
  const float* ln_b[3]  = {(const float*)d_in[19], (const float*)d_in[20], (const float*)d_in[21]};

  // workspace layout (~82 MB)
  float*    h     = (float*)d_ws;                         // [N,128]
  float*    lin   = h     + (size_t)NNODES * HID;         // [N,512] max
  float*    agg   = lin   + (size_t)NNODES * 512;         // [N,128]
  float*    a_s   = agg   + (size_t)NNODES * HID;         // [N,4]
  float*    a_d   = a_s   + (size_t)NNODES * HEADS;       // [N,4]
  unsigned* amax  = (unsigned*)(a_d + (size_t)NNODES * HEADS); // [N,4]
  float*    denom = (float*)(amax + (size_t)NNODES * HEADS);   // [N,4]
  float*    ex    = denom + (size_t)NNODES * HEADS;       // [ETOT,4]
  bf16*     abf   = (bf16*)(ex + (size_t)ETOT * HEADS);   // [N,256] bf16 (x)
  bf16*     wbf   = abf + (size_t)NNODES * 256;           // [512,128] bf16 max
  bf16*     hbf   = wbf + (size_t)512 * 128;              // [N,128] bf16 activations

  // ---- input projection: h = relu(x @ proj_w + proj_b); also emit bf16 h
  k_cvt_bf16 <<<cdiv(NNODES * 256, 256), 256, 0, stream>>>(x, abf, NNODES * 256);
  k_tcvt_bf16<<<cdiv(256 * 128, 256),    256, 0, stream>>>(proj_w, wbf, 256, 128);
  {
    int waves = (NNODES / 16) * (128 / 64);
    k_wmma_gemm<<<cdiv(waves, 4), 128, 0, stream>>>(abf, wbf, proj_b, h, hbf,
                                                    NNODES, 256, 128, 1);
  }

  for (int l = 0; l < 3; ++l) {
    const int C    = (l < 2) ? 32 : 128;
    const int Nout = HEADS * C;   // 128 or 512

    // lin = h @ lin_w[l]  (A is the fused bf16 activation copy)
    k_tcvt_bf16<<<cdiv(HID * Nout, 256), 256, 0, stream>>>(lin_w[l], wbf, HID, Nout);
    int waves = (NNODES / 16) * (Nout / 64);
    k_wmma_gemm<<<cdiv(waves, 4), 128, 0, stream>>>(hbf, wbf, nullptr, lin, nullptr,
                                                    NNODES, HID, Nout, 0);

    // attention scores + zeroed reduction buffers
    k_att_scores<<<cdiv(NNODES * HEADS, 256), 256, 0, stream>>>(lin, att_s[l], att_d[l], a_s, a_d, C);
    k_zero<<<cdiv(NNODES * HID, 256), 256, 0, stream>>>(agg, NNODES * HID);
    k_zero<<<cdiv(NNODES * HEADS * 2, 256), 256, 0, stream>>>((float*)amax, NNODES * HEADS * 2); // amax+denom

    // segment softmax + scatter aggregation
    k_edge_max<<<cdiv(ETOT, 256), 256, 0, stream>>>(ei, a_s, a_d, amax);
    k_edge_exp<<<cdiv(ETOT, 256), 256, 0, stream>>>(ei, a_s, a_d, amax, denom, ex);
    if (l < 2)
      k_edge_aggr_concat<<<cdiv(ETOT * 128, 256), 256, 0, stream>>>(ei, ex, denom, lin, agg);
    else
      k_edge_aggr_mean  <<<cdiv(ETOT * 128, 256), 256, 0, stream>>>(ei, ex, denom, lin, agg);

    // bias + layernorm (+relu for l<2) (+residual for l>0); layer 2 writes d_out
    float* hout = (l == 2) ? (float*)d_out : h;
    bf16*  hbfo = (l == 2) ? nullptr : hbf;
    k_finalize<<<NNODES / 8, 256, 0, stream>>>(agg, bias[l], ln_s[l], ln_b[l], h, hout, hbfo,
                                               (l < 2) ? 1 : 0, (l > 0) ? 1 : 0);
  }
  (void)in_sizes; (void)n_in; (void)out_size; (void)ws_size;
}
